// BondOrderModel_50680614092953
// MI455X (gfx1250) — matrix operand order repaired and checked by reading, not compiled, not run on hardware
//
#include <hip/hip_runtime.h>

// ---------------------------------------------------------------------------
// MACE-like GNN forward, reduced: only the l=0 channel of node_feats ever
// reaches the outputs, so messages collapse to m = w_e * h[sender] (E x 64).
// Heavy op: per-iteration edge GEMM (E x 128)@(128 x 64) done with
// v_wmma_f32_16x16x32_bf16, edge features (rad x sh) rebuilt in registers.
// B operands kept in LDS in fragment-major (swizzled) layout so each WMMA
// B-operand is two contiguous ds_load_b128 -- no VGPR spills.
// ---------------------------------------------------------------------------

typedef __attribute__((ext_vector_type(16))) __bf16 v16bf;
typedef __attribute__((ext_vector_type(8)))  float  v8f;

#define WMMA_BF16(A, B, C) \
  __builtin_amdgcn_wmma_f32_16x16x32_bf16(false, (A), false, (B), (short)0, (C), false, false)

namespace {
constexpr int N_   = 10000;
constexpr int E_   = 160000;
constexpr int CIN_ = 10;
constexpr int C_   = 64;
constexpr int NB_  = 8;
constexpr int NSH_ = 16;
constexpr int K_   = NB_ * NSH_;   // 128
constexpr int NG_  = 16;
constexpr int EDGE_TILES = E_ / 16;   // 10000
constexpr int NODE_TILES = N_ / 16;   // 625
}

// ---------------------------------------------------------------------------
__global__ void zero_kernel(float* __restrict__ p, int n) {
  int i = blockIdx.x * blockDim.x + threadIdx.x;
  if (i < n) p[i] = 0.0f;
}

// h0 = attrs @ W_node ; e0 = attrs @ atomic_energies -> graph energy row 0
__global__ void node_init_kernel(const float* __restrict__ attrs,
                                 const float* __restrict__ Wnode,
                                 const float* __restrict__ aeng,
                                 const int*   __restrict__ batch,
                                 float* __restrict__ h,
                                 float* __restrict__ gen) {
  int node = blockIdx.x * 4 + (threadIdx.x >> 6);
  int c    = threadIdx.x & 63;
  if (node >= N_) return;
  const float* ar = attrs + node * CIN_;
  float s = 0.0f;
#pragma unroll
  for (int ci = 0; ci < CIN_; ++ci) s += ar[ci] * Wnode[ci * C_ + c];
  h[node * C_ + c] = s;
  if (c == 0) {
    float e = 0.0f;
#pragma unroll
    for (int ci = 0; ci < CIN_; ++ci) e += ar[ci] * aeng[ci];
    atomicAdd(&gen[batch[node]], e);
  }
}

// ---------------------------------------------------------------------------
// Edge kernel. B fragment (kt,nt,lane): element t <-> K = 32*kt + 8*(lane>>4)
// + (t&7) + 16*(t>>3), col = 16*nt + (lane&15). Stored fragment-major in LDS: bf16 index
// ((kt*4+nt)*32 + lane)*16 + t, so each lane's fragment is 32 contiguous bytes.
// ---------------------------------------------------------------------------
__global__ __launch_bounds__(256, 1)
void edge_wmma_kernel(const float* __restrict__ evec,
                      const float* __restrict__ elen,
                      const int*   __restrict__ eidx,
                      const float* __restrict__ Wr,      // 128x64, this iter
                      const float* __restrict__ h,
                      float* __restrict__ agg) {
  __shared__ v16bf Bfrag[16 * 32];   // 16 KB, [kt*4+nt][lane]
  {
    __bf16* bp = (__bf16*)Bfrag;
    for (int idx = threadIdx.x; idx < K_ * C_; idx += blockDim.x) {
      int t    = idx & 15;
      int lane = (idx >> 4) & 31;
      int f    = idx >> 9;          // kt*4 + nt
      int nt   = f & 3;
      int kt   = f >> 2;
      int Kk   = 32 * kt + 8 * (lane >> 4) + (t & 7) + 16 * (t >> 3);
      int cn   = 16 * nt + (lane & 15);
      bp[idx] = (__bf16)Wr[Kk * C_ + cn];
    }
  }
  __syncthreads();

  const int lane = threadIdx.x & 31;
  const int l15  = lane & 15;
  const int hi   = lane >> 4;
  const int gwave  = blockIdx.x * (blockDim.x >> 5) + (threadIdx.x >> 5);
  const int nwaves = gridDim.x * (blockDim.x >> 5);

  for (int tile = gwave; tile < EDGE_TILES; tile += nwaves) {
    const int base = tile * 16;

    // --- per-lane A-row geometry (row = l15; both lane halves need it) ---
    const int eA = base + l15;
    float ex = evec[3 * eA], ey = evec[3 * eA + 1], ez = evec[3 * eA + 2];
    float r  = elen[eA];
    float inv = 1.0f / r;
    float x = ex * inv, y = ey * inv, z = ez * inv;

    float q[16];
    q[0] = 1.0f;            q[1] = x;               q[2] = y;      q[3] = z;
    q[4] = x * y;           q[5] = y * z;           q[6] = 3.0f * z * z - 1.0f;
    q[7] = x * z;           q[8] = x * x - y * y;
    q[9] = y * (3.0f * x * x - y * y);
    q[10] = x * y * z;
    q[11] = y * (5.0f * z * z - 1.0f);
    q[12] = z * (5.0f * z * z - 3.0f);
    q[13] = x * (5.0f * z * z - 1.0f);
    q[14] = z * (x * x - y * y);
    q[15] = x * (x * x - 3.0f * y * y);

    // radial: bessel * polynomial cutoff envelope (p = 6)
    float u  = r * 0.2f;
    float u2 = u * u;
    float u6 = u2 * u2 * u2;
    float env = 1.0f - 28.0f * u6 + 48.0f * u6 * u - 21.0f * u6 * u2;
    env = (u < 1.0f) ? env : 0.0f;
    float pref = 0.632455532f * inv * env;   // sqrt(2/5)/r * env
    float rad[8];
#pragma unroll
    for (int b = 0; b < 8; ++b)
      rad[b] = pref * __sinf((float)(b + 1) * 0.6283185307f * r);

    // lane-half K selection: sv[j] = sh[8*hi + j]
    float sv[8];
#pragma unroll
    for (int j = 0; j < 8; ++j) sv[j] = hi ? q[8 + j] : q[j];

    v8f acc[4] = {};
#pragma unroll
    for (int kt = 0; kt < 4; ++kt) {
      v16bf a;
#pragma unroll
      for (int j = 0; j < 8; ++j) {
        // global k1 = 32*kt + 8*hi + j  -> rad[2*kt]   * sh[8*hi+j]
        // global k2 = k1 + 16          -> rad[2*kt+1] * sh[8*hi+j]
        a[j]     = (__bf16)(rad[2 * kt]     * sv[j]);
        a[j + 8] = (__bf16)(rad[2 * kt + 1] * sv[j]);
      }
#pragma unroll
      for (int nt = 0; nt < 4; ++nt)
        acc[nt] = WMMA_BF16(a, Bfrag[(kt * 4 + nt) * 32 + lane], acc[nt]);
    }

    // --- fused epilogue: D row = r8 + 8*hi, col = 16*nt + l15 ---
    int sidx[8], ridx[8];
#pragma unroll
    for (int r8 = 0; r8 < 8; ++r8) {
      int e = base + r8 + 8 * hi;
      sidx[r8] = eidx[e];        // sender
      ridx[r8] = eidx[E_ + e];   // receiver
    }
#pragma unroll
    for (int nt = 0; nt < 4; ++nt) {
      int ch = 16 * nt + l15;
#pragma unroll
      for (int r8 = 0; r8 < 8; ++r8) {
        float m = acc[nt][r8] * h[sidx[r8] * C_ + ch];
        atomicAdd(&agg[ridx[r8] * C_ + ch], m);
      }
    }
  }
}

// ---------------------------------------------------------------------------
// Mix kernel: h = agg @ W_mix + attrs @ W_sc (WMMA for the 64x64 mix),
// fused readout energy = h @ w_ro accumulated per graph.
// ---------------------------------------------------------------------------
__global__ __launch_bounds__(256, 1)
void mix_wmma_kernel(const float* __restrict__ agg,
                     const float* __restrict__ Wmix,  // 64x64, this iter
                     const float* __restrict__ Wsc,   // 10x64, this iter
                     const float* __restrict__ wro,   // 64,    this iter
                     const float* __restrict__ attrs,
                     const int*   __restrict__ batch,
                     float* __restrict__ h,
                     float* __restrict__ gen,
                     int iter) {
  __shared__ v16bf Bfrag[8 * 32];    // 8 KB, [kt*4+nt][lane]
  __shared__ float Ws[CIN_ * C_];    // 2.5 KB
  {
    __bf16* bp = (__bf16*)Bfrag;
    for (int idx = threadIdx.x; idx < C_ * C_; idx += blockDim.x) {
      int t    = idx & 15;
      int lane = (idx >> 4) & 31;
      int f    = idx >> 9;          // kt*4 + nt  (kt in 0..1)
      int nt   = f & 3;
      int kt   = f >> 2;
      int Kk   = 32 * kt + 8 * (lane >> 4) + (t & 7) + 16 * (t >> 3);
      int cn   = 16 * nt + (lane & 15);
      bp[idx] = (__bf16)Wmix[Kk * C_ + cn];
    }
  }
  for (int j = threadIdx.x; j < CIN_ * C_; j += blockDim.x) Ws[j] = Wsc[j];
  __syncthreads();

  const int lane = threadIdx.x & 31;
  const int l15  = lane & 15;
  const int hi   = lane >> 4;
  const int gwave  = blockIdx.x * (blockDim.x >> 5) + (threadIdx.x >> 5);
  const int nwaves = gridDim.x * (blockDim.x >> 5);

  for (int tile = gwave; tile < NODE_TILES; tile += nwaves) {
    const int base = tile * 16;
    const float* aggRow = agg + (base + l15) * C_;   // A row = l15

    v8f acc[4] = {};
#pragma unroll
    for (int kt = 0; kt < 2; ++kt) {
      v16bf a;
#pragma unroll
      for (int j = 0; j < 8; ++j) {
        int k = 32 * kt + 8 * hi + j;
        a[j]     = (__bf16)aggRow[k];
        a[j + 8] = (__bf16)aggRow[k + 16];
      }
#pragma unroll
      for (int nt = 0; nt < 4; ++nt)
        acc[nt] = WMMA_BF16(a, Bfrag[(kt * 4 + nt) * 32 + lane], acc[nt]);
    }

    // epilogue: D row = r8 + 8*hi, col = 16*nt + l15
#pragma unroll
    for (int r8 = 0; r8 < 8; ++r8) {
      const int node = base + r8 + 8 * hi;
      const float* ar = attrs + node * CIN_;
      float part = 0.0f;
#pragma unroll
      for (int nt = 0; nt < 4; ++nt) {
        int ch = 16 * nt + l15;
        float skip = 0.0f;
#pragma unroll
        for (int ci = 0; ci < CIN_; ++ci)
          skip += ar[ci] * Ws[ci * C_ + ch];
        float val = acc[nt][r8] + skip;
        h[node * C_ + ch] = val;
        part += val * wro[ch];
      }
      // 16-lane reduce: each half-wave owns one node row
      part += __shfl_xor(part, 8, 16);
      part += __shfl_xor(part, 4, 16);
      part += __shfl_xor(part, 2, 16);
      part += __shfl_xor(part, 1, 16);
      if (l15 == 0)
        atomicAdd(&gen[(iter + 1) * NG_ + batch[node]], part);
    }
  }
}

// ---------------------------------------------------------------------------
__global__ void finalize_kernel(const float* __restrict__ gen,
                                float* __restrict__ out) {
  int t = threadIdx.x;
  if (t < 4 * NG_) out[NG_ + t] = gen[t];         // energies (4,16)
  if (t < NG_) {
    float s = 0.0f;
#pragma unroll
    for (int i = 0; i < 4; ++i) s += gen[i * NG_ + t];
    out[t] = s;                                    // total_energy (16,)
  }
}

// ---------------------------------------------------------------------------
extern "C" void kernel_launch(void* const* d_in, const int* in_sizes, int n_in,
                              void* d_out, int out_size, void* d_ws, size_t ws_size,
                              hipStream_t stream) {
  const float* attrs = (const float*)d_in[0];
  const float* evec  = (const float*)d_in[1];
  const float* elen  = (const float*)d_in[2];
  const int*   eidx  = (const int*)  d_in[3];
  const int*   batch = (const int*)  d_in[4];
  const float* aeng  = (const float*)d_in[6];
  const float* Wnode = (const float*)d_in[7];
  const float* Wr    = (const float*)d_in[8];
  const float* Wmix  = (const float*)d_in[9];
  const float* Wsc   = (const float*)d_in[10];
  const float* wro   = (const float*)d_in[11];
  float* out = (float*)d_out;

  float* h   = (float*)d_ws;       // N*64
  float* agg = h + N_ * C_;        // N*64
  float* gen = agg + N_ * C_;      // 4*16 graph energies

  zero_kernel<<<1, 64, 0, stream>>>(gen, 4 * NG_);
  node_init_kernel<<<(N_ + 3) / 4, 256, 0, stream>>>(attrs, Wnode, aeng, batch, h, gen);

  for (int i = 0; i < 3; ++i) {
    zero_kernel<<<(N_ * C_ + 255) / 256, 256, 0, stream>>>(agg, N_ * C_);
    edge_wmma_kernel<<<320, 256, 0, stream>>>(evec, elen, eidx,
                                              Wr + i * K_ * C_, h, agg);
    mix_wmma_kernel<<<80, 256, 0, stream>>>(agg, Wmix + i * C_ * C_,
                                            Wsc + i * CIN_ * C_, wro + i * C_,
                                            attrs, batch, h, gen, i);
  }
  finalize_kernel<<<1, 64, 0, stream>>>(gen, out);
}
